// SurfaceNet_63763084477189
// MI455X (gfx1250) — matrix operand order repaired and checked by reading, not compiled, not run on hardware
//
#include <hip/hip_runtime.h>

// ---------------------------------------------------------------------------
// SurfaceNet GNN forward on gfx1250 (MI455X), wave32 + WMMA bf16.
//   h   = relu(x @ enc_w + b)                               [100k,128]
//   agg = scatter_mean(h[src] * (edge_attr@e_w+b), dst)     per conv
//   h   = relu(concat(agg, h) @ l_w + b)                    [100k,256] / [100k,512]
//   out = relu(h) @ dec_w + dec_b                           [100k,2]
// Activations kept in bf16 (L2-resident), GEMMs via v_wmma_f32_16x16x32_bf16,
// scatter via fp32 global atomics.
// ---------------------------------------------------------------------------

#define N_NODES 100000
#define N_EDGES 1000000
#define IN_FEAT 32
#define ENC     128
#define C0      256
#define C1      512

typedef __bf16 v16bf __attribute__((ext_vector_type(16)));
typedef __bf16 v8bf  __attribute__((ext_vector_type(8)));
typedef float  v8f   __attribute__((ext_vector_type(8)));
typedef unsigned short v4us __attribute__((ext_vector_type(4)));

__device__ __forceinline__ float bf2f(unsigned short u) {
    unsigned int x = ((unsigned int)u) << 16;
    return __builtin_bit_cast(float, x);
}
__device__ __forceinline__ unsigned short f2bf(float f) {
    unsigned int u = __builtin_bit_cast(unsigned int, f);
    u += 0x7FFFu + ((u >> 16) & 1u);   // round-to-nearest-even
    return (unsigned short)(u >> 16);
}

// --- fp32 -> bf16 elementwise convert ---------------------------------------
__global__ void k_convert_bf16(const float* __restrict__ in,
                               unsigned short* __restrict__ out, int n) {
    int t = blockIdx.x * blockDim.x + threadIdx.x;
    if (t < n) out[t] = f2bf(in[t]);
}

// --- weight transpose+convert: W[K,N] fp32 -> Wt[N,K] bf16 ------------------
__global__ void k_transpose_w(const float* __restrict__ W,
                              unsigned short* __restrict__ Wt, int K, int N) {
    int t = blockIdx.x * blockDim.x + threadIdx.x;
    if (t >= K * N) return;
    int k = t / N, n = t % N;          // coalesced read of W
    Wt[(size_t)n * K + k] = f2bf(W[t]);
}

// --- bf16 GEMM with f32 accumulate via WMMA ---------------------------------
// One wave computes a 16x64 output strip (4 x 16x16 WMMA tiles, A reused 4x).
// A: [M,K] bf16 row-major (lda). Bt: [N,K] bf16 (pre-transposed weights).
// Epilogue: +bias, optional relu, bf16 store into C[row*ldc + coff + col].
#define GEMM_WAVES 8
__global__ __launch_bounds__(256)
void k_gemm_bf16(const unsigned short* __restrict__ A, int lda,
                 const unsigned short* __restrict__ Bt,
                 const float* __restrict__ bias,
                 unsigned short* __restrict__ C, int ldc, int coff,
                 int M, int N, int K, int relu) {
    const int lane = threadIdx.x & 31;
    const int wave = threadIdx.x >> 5;
    const int tilesN = N >> 6;                       // 64 cols per wave
    const int tile = blockIdx.x * GEMM_WAVES + wave;
    if (tile >= (M >> 4) * tilesN) return;
    const int tm = (tile / tilesN) << 4;
    const int tn = (tile % tilesN) << 6;

    const int half = lane >> 4;                      // 0: lanes 0-15, 1: 16-31
    const int l16  = lane & 15;
    // A-matrix 16x32 layout (contiguous, matches B): lanes 0-15 hold K = 0..15,
    // lanes 16-31 hold K = 16..31 of row M = lane%16.
    const int ka0 = half ? 16 : 0;
    const int ka1 = half ? 24 : 8;
    const unsigned short* Arow = A + (size_t)(tm + l16) * lda;

    v8f acc[4] = {};
    for (int k0 = 0; k0 < K; k0 += 32) {
        v16bf a;
        {
            v8bf alo = *(const v8bf*)(Arow + k0 + ka0);
            v8bf ahi = *(const v8bf*)(Arow + k0 + ka1);
#pragma unroll
            for (int i = 0; i < 8; ++i) { a[i] = alo[i]; a[i + 8] = ahi[i]; }
        }
        // 16-bit B-matrix 32x16 (contiguous k order, same as A): lanes 0-15 hold K=0..15, lanes 16-31 hold
        // K=16..31 of column N = lane%16 -> contiguous in [N,K] storage.
        const int kb = k0 + (half << 4);
#pragma unroll
        for (int nt = 0; nt < 4; ++nt) {
            v16bf b = *(const v16bf*)(Bt + (size_t)(tn + nt * 16 + l16) * K + kb);
            acc[nt] = __builtin_amdgcn_wmma_f32_16x16x32_bf16(
                false, a, false, b, (short)0, acc[nt], false, false);
        }
    }
    // C/D layout: VGPR r holds (row = tm + r + 8*half, col = tn + nt*16 + l16)
#pragma unroll
    for (int nt = 0; nt < 4; ++nt) {
        const int col = tn + nt * 16 + l16;
        const float bv = bias[col];
#pragma unroll
        for (int r = 0; r < 8; ++r) {
            const int row = tm + r + (half << 3);
            float v = acc[nt][r] + bv;
            if (relu) v = v > 0.0f ? v : 0.0f;
            C[(size_t)row * ldc + coff + col] = f2bf(v);
        }
    }
}

// --- edge message + scatter-add ---------------------------------------------
// thread handles 4 features of one edge; F=128 -> 1 wave/edge, F=256 -> 2.
__global__ __launch_bounds__(256)
void k_msg_scatter(const int* __restrict__ src, const int* __restrict__ dst,
                   const float* __restrict__ edge_attr,
                   const unsigned short* __restrict__ h, int ldh, int hoff,
                   const float* __restrict__ ew, const float* __restrict__ eb,
                   float* __restrict__ s, float* __restrict__ deg,
                   int E, int F) {
    const int chunks = F >> 2;
    long long t = (long long)blockIdx.x * blockDim.x + threadIdx.x;
    if (t >= (long long)E * chunks) return;
    const int e  = (int)(t / chunks);
    const int fi = ((int)(t % chunks)) << 2;
    const int se = src[e], de = dst[e];
    const float a0 = edge_attr[2 * e], a1 = edge_attr[2 * e + 1];
    v4us hv = *(const v4us*)(h + (size_t)se * ldh + hoff + fi);
    float* so = s + (size_t)de * F + fi;
#pragma unroll
    for (int j = 0; j < 4; ++j) {
        float ea = fmaf(a0, ew[fi + j], fmaf(a1, ew[F + fi + j], eb[fi + j]));
        float m  = bf2f(hv[j]) * ea;
        __hip_atomic_fetch_add(so + j, m, __ATOMIC_RELAXED,
                               __HIP_MEMORY_SCOPE_AGENT);
    }
    if (deg != nullptr && fi == 0)
        __hip_atomic_fetch_add(deg + de, 1.0f, __ATOMIC_RELAXED,
                               __HIP_MEMORY_SCOPE_AGENT);
}

// --- scatter-mean normalize -> bf16 into concat buffer ----------------------
__global__ void k_normalize(const float* __restrict__ s,
                            const float* __restrict__ deg,
                            unsigned short* __restrict__ cat,
                            int ldcat, int coff, int nnodes, int F) {
    long long t = (long long)blockIdx.x * blockDim.x + threadIdx.x;
    if (t >= (long long)nnodes * F) return;
    const int i = (int)(t / F), f = (int)(t % F);
    float d = deg[i];
    d = d > 1.0f ? d : 1.0f;
    cat[(size_t)i * ldcat + coff + f] = f2bf(s[t] / d);
}

// --- decoder: [100k,512] bf16 @ [512,2] fp32 -> fp32 out --------------------
__global__ __launch_bounds__(256)
void k_decoder(const unsigned short* __restrict__ h2,
               const float* __restrict__ dw, const float* __restrict__ db,
               float* __restrict__ out, int M) {
    const int wid  = (int)((blockIdx.x * (long long)blockDim.x + threadIdx.x) >> 5);
    const int lane = threadIdx.x & 31;
    if (wid >= M) return;
    const unsigned short* hp = h2 + (size_t)wid * C1 + lane * 16;
    float c0 = 0.0f, c1 = 0.0f;
#pragma unroll
    for (int j = 0; j < 16; ++j) {
        const float hv = bf2f(hp[j]);
        const int k = lane * 16 + j;
        c0 = fmaf(hv, dw[k * 2 + 0], c0);
        c1 = fmaf(hv, dw[k * 2 + 1], c1);
    }
#pragma unroll
    for (int off = 16; off > 0; off >>= 1) {
        c0 += __shfl_down(c0, off, 32);
        c1 += __shfl_down(c1, off, 32);
    }
    if (lane == 0) {
        out[(size_t)wid * 2 + 0] = c0 + db[0];
        out[(size_t)wid * 2 + 1] = c1 + db[1];
    }
}

static inline unsigned cdiv(long long a, long long b) {
    return (unsigned)((a + b - 1) / b);
}

extern "C" void kernel_launch(void* const* d_in, const int* in_sizes, int n_in,
                              void* d_out, int out_size, void* d_ws, size_t ws_size,
                              hipStream_t stream) {
    const float* x     = (const float*)d_in[0];
    const float* eattr = (const float*)d_in[1];
    const int*   eidx  = (const int*)d_in[2];
    const float* enc_w = (const float*)d_in[3];
    const float* enc_b = (const float*)d_in[4];
    const float* e_w0  = (const float*)d_in[5];
    const float* e_b0  = (const float*)d_in[6];
    const float* l_w0  = (const float*)d_in[7];
    const float* l_b0  = (const float*)d_in[8];
    const float* e_w1  = (const float*)d_in[9];
    const float* e_b1  = (const float*)d_in[10];
    const float* l_w1  = (const float*)d_in[11];
    const float* l_b1  = (const float*)d_in[12];
    const float* dec_w = (const float*)d_in[13];
    const float* dec_b = (const float*)d_in[14];
    const int* src = eidx;
    const int* dst = eidx + N_EDGES;
    float* out = (float*)d_out;

    // workspace carve-out (256B aligned)
    char* p = (char*)d_ws;
    auto carve = [&](size_t bytes) -> char* {
        char* r = p;
        p += (bytes + 255) & ~(size_t)255;
        return r;
    };
    unsigned short* xb   = (unsigned short*)carve((size_t)N_NODES * IN_FEAT * 2);
    unsigned short* wenc = (unsigned short*)carve((size_t)ENC * IN_FEAT * 2); // [N,K]
    unsigned short* wl0  = (unsigned short*)carve((size_t)C0 * C0 * 2);
    unsigned short* wl1  = (unsigned short*)carve((size_t)C1 * C1 * 2);
    unsigned short* cat0 = (unsigned short*)carve((size_t)N_NODES * C0 * 2);  // [agg0|h_enc]
    unsigned short* cat1 = (unsigned short*)carve((size_t)N_NODES * C1 * 2);  // [agg1|h1]
    unsigned short* h2   = (unsigned short*)carve((size_t)N_NODES * C1 * 2);
    float*          sacc = (float*)carve((size_t)N_NODES * C0 * 4);           // reused
    float*          deg  = (float*)carve((size_t)N_NODES * 4);
    (void)ws_size; (void)in_sizes; (void)n_in; (void)out_size;

    const int T = 256;

    // 0) convert inputs / weights to bf16 (weights transposed to [N,K])
    k_convert_bf16<<<cdiv((long long)N_NODES * IN_FEAT, T), T, 0, stream>>>(
        x, xb, N_NODES * IN_FEAT);
    k_transpose_w<<<cdiv(IN_FEAT * ENC, T), T, 0, stream>>>(enc_w, wenc, IN_FEAT, ENC);
    k_transpose_w<<<cdiv(C0 * C0, T), T, 0, stream>>>(l_w0, wl0, C0, C0);
    k_transpose_w<<<cdiv(C1 * C1, T), T, 0, stream>>>(l_w1, wl1, C1, C1);

    // 1) encoder: h_enc = relu(x @ enc_w + b) -> cat0 cols [128..255]
    k_gemm_bf16<<<cdiv((long long)(N_NODES / 16) * (ENC / 64), GEMM_WAVES),
                  T, 0, stream>>>(xb, IN_FEAT, wenc, enc_b,
                                  cat0, C0, ENC, N_NODES, ENC, IN_FEAT, 1);

    // 2) conv0 edge phase: scatter-add msgs + degree
    hipMemsetAsync(sacc, 0, (size_t)N_NODES * ENC * 4, stream);
    hipMemsetAsync(deg, 0, (size_t)N_NODES * 4, stream);
    k_msg_scatter<<<cdiv((long long)N_EDGES * (ENC / 4), T), T, 0, stream>>>(
        src, dst, eattr, cat0, C0, ENC, e_w0, e_b0, sacc, deg, N_EDGES, ENC);
    k_normalize<<<cdiv((long long)N_NODES * ENC, T), T, 0, stream>>>(
        sacc, deg, cat0, C0, 0, N_NODES, ENC);

    // 3) conv0 GEMM: h1 = relu(cat0 @ l_w0 + b) -> cat1 cols [256..511]
    k_gemm_bf16<<<cdiv((long long)(N_NODES / 16) * (C0 / 64), GEMM_WAVES),
                  T, 0, stream>>>(cat0, C0, wl0, l_b0,
                                  cat1, C1, C0, N_NODES, C0, C0, 1);

    // 4) conv1 edge phase
    hipMemsetAsync(sacc, 0, (size_t)N_NODES * C0 * 4, stream);
    k_msg_scatter<<<cdiv((long long)N_EDGES * (C0 / 4), T), T, 0, stream>>>(
        src, dst, eattr, cat1, C1, C0, e_w1, e_b1, sacc, nullptr, N_EDGES, C0);
    k_normalize<<<cdiv((long long)N_NODES * C0, T), T, 0, stream>>>(
        sacc, deg, cat1, C1, 0, N_NODES, C0);

    // 5) conv1 GEMM: h2 = relu(cat1 @ l_w1 + b)
    k_gemm_bf16<<<cdiv((long long)(N_NODES / 16) * (C1 / 64), GEMM_WAVES),
                  T, 0, stream>>>(cat1, C1, wl1, l_b1,
                                  h2, C1, 0, N_NODES, C1, C1, 1);

    // 6) decoder
    k_decoder<<<cdiv((long long)N_NODES * 32, T), T, 0, stream>>>(
        h2, dec_w, dec_b, out, N_NODES);
}
